// TransformerBlock_3633542332823
// MI455X (gfx1250) — matrix operand order repaired and checked by reading, not compiled, not run on hardware
//
#include <hip/hip_runtime.h>
#include <hip/hip_bf16.h>
#include <math.h>

typedef __bf16 bf16;
typedef __attribute__((ext_vector_type(16))) __bf16 v16bf;
typedef __attribute__((ext_vector_type(8)))  float  v8f;

#define HW_   65536
#define NPIX_ 262144   // B*H*W
#define IMGW  256

__device__ __forceinline__ v8f wmma_bf16(v16bf a, v16bf b, v8f c) {
  return __builtin_amdgcn_wmma_f32_16x16x32_bf16(false, a, false, b, (short)0, c, false, false);
}

// A fragment: A is MxK = 16x32 bf16. lane L holds row m = L&15.
// elements 0..15 = K (0..15)+16*(L>=16)
__device__ __forceinline__ v16bf load_a_frag(const bf16* W, int ld, int m0, int k0, int lane) {
  const bf16* row = W + (m0 + (lane & 15)) * ld + k0 + (((lane >> 4) & 1) << 4);
  v16bf a;
#pragma unroll
  for (int e = 0; e < 16; ++e) a[e] = row[e];
  return a;
}

// B fragment: B is KxN = 32x16 bf16, stored pixel-major X[n][k] (k contiguous).
// lane L holds col n = L&15; elements 0..15 = K (0..15)+16*(L>=16)
__device__ __forceinline__ v16bf load_b_frag(const bf16* X, int ld, int n0, int k0, int lane) {
  const bf16* col = X + (n0 + (lane & 15)) * ld + k0 + (((lane >> 4) & 1) << 4);
  v16bf b;
#pragma unroll
  for (int e = 0; e < 16; ++e) b[e] = col[e];
  return b;
}

// ---------------------------------------------------------------------------
// Kernel 1/3: LayerNorm(C=64) + conv1x1 GEMM (OCT*16 output channels) -> bf16
// out layout: [oc][NPIX]. 256 pixels per block, 256 threads (8 waves).
// ---------------------------------------------------------------------------
template <int OCT>
__global__ void __launch_bounds__(256) ln_conv1x1_kernel(
    const float* __restrict__ x, const float* __restrict__ lnw,
    const float* __restrict__ lnb, const float* __restrict__ Wg,
    int oc_real, bf16* __restrict__ out) {
  __shared__ bf16 Wl[OCT * 16][64];
  __shared__ bf16 xn[256][72];
  const int tid = threadIdx.x;

  // stage weights (zero-padded rows beyond oc_real)
  for (int i = tid; i < OCT * 16 * 64; i += 256) {
    int o = i >> 6, k = i & 63;
    Wl[o][k] = (o < oc_real) ? (bf16)Wg[o * 64 + k] : (bf16)0.f;
  }

  const int P0 = blockIdx.x * 256;
  const float* xb = x + (size_t)(P0 >> 16) * (64 * HW_) + (P0 & (HW_ - 1));

  // LayerNorm: one pixel per thread (coalesced across threads per channel)
  {
    const int p = tid;
    float s = 0.f, ss = 0.f;
#pragma unroll 8
    for (int c = 0; c < 64; ++c) { float v = xb[c * HW_ + p]; s += v; ss += v * v; }
    float mu  = s * (1.f / 64.f);
    float var = ss * (1.f / 64.f) - mu * mu;
    float r   = rsqrtf(var + 1e-5f);
#pragma unroll 8
    for (int c = 0; c < 64; ++c) {
      float v = (xb[c * HW_ + p] - mu) * r * lnw[c] + lnb[c];
      xn[p][c] = (bf16)v;
    }
  }
  __syncthreads();

  const int lane = tid & 31, wave = tid >> 5;
  for (int sl = wave; sl < 16; sl += 8) {
    const int n0 = sl * 16;
    for (int mt = 0; mt < OCT; ++mt) {
      v8f acc = {};
#pragma unroll
      for (int kt = 0; kt < 2; ++kt) {
        v16bf a = load_a_frag(&Wl[0][0], 64, mt * 16, kt * 32, lane);
        v16bf b = load_b_frag(&xn[0][0], 72, n0, kt * 32, lane);
        acc = wmma_bf16(a, b, acc);
      }
      const int n  = n0 + (lane & 15);
      const int mb = mt * 16 + ((lane >> 4) << 3);
#pragma unroll
      for (int r = 0; r < 8; ++r) {
        int o = mb + r;
        if (o < oc_real) out[(size_t)o * NPIX_ + P0 + n] = (bf16)acc[r];
      }
    }
  }
}

// ---------------------------------------------------------------------------
// Kernel 2: dwconv3x3(qkv) + windowed channel-attention + proj + residual
// One block per 8x8 window. 256 threads.
// ---------------------------------------------------------------------------
__global__ void __launch_bounds__(256) attn_kernel(
    const bf16* __restrict__ qkv, const float* __restrict__ dww,
    const float* __restrict__ temp, const float* __restrict__ projw,
    const float* __restrict__ projb, const float* __restrict__ x,
    float* __restrict__ x2) {
  __shared__ bf16  qh[192][100];   // 10x10 halo, bf16
  __shared__ float qd[192][64];    // dwconv result (q|k|v), fp32
  __shared__ float attn[2][32][32];
  __shared__ bf16  ofrag[64][72];  // attention output, pixel-major [pix][ch]
  __shared__ bf16  Wp[64][64];     // proj weights

  const int tid = threadIdx.x;
  const int wi = blockIdx.x;
  const int b = wi >> 10, win = wi & 1023;
  const int h0 = (win >> 5) * 8, w0 = (win & 31) * 8;

  // halo load (zero-padded SAME borders)
  for (int i = tid; i < 192 * 100; i += 256) {
    int c = i / 100, pos = i % 100;
    int gh = h0 + pos / 10 - 1, gw = w0 + pos % 10 - 1;
    bf16 v = (bf16)0.f;
    if ((unsigned)gh < 256u && (unsigned)gw < 256u)
      v = qkv[(size_t)c * NPIX_ + b * HW_ + gh * IMGW + gw];
    qh[c][pos] = v;
  }
  for (int i = tid; i < 64 * 64; i += 256) Wp[i >> 6][i & 63] = (bf16)projw[i];
  __syncthreads();

  // depthwise 3x3
  for (int i = tid; i < 192 * 64; i += 256) {
    int c = i >> 6, p = i & 63, iy = p >> 3, ix = p & 7;
    float acc = 0.f;
#pragma unroll
    for (int ky = 0; ky < 3; ++ky)
#pragma unroll
      for (int kx = 0; kx < 3; ++kx)
        acc += (float)qh[c][(iy + ky) * 10 + ix + kx] * dww[c * 9 + ky * 3 + kx];
    qd[c][p] = acc;
  }
  __syncthreads();

  // l2-normalize rows of q (ch 0..63) and k (ch 64..127) over d=64 pixels
  if (tid < 128) {
    float* row = qd[tid];
    float ss = 0.f;
#pragma unroll 8
    for (int p = 0; p < 64; ++p) ss += row[p] * row[p];
    float sc = 1.f / fmaxf(sqrtf(ss), 1e-12f);
#pragma unroll 8
    for (int p = 0; p < 64; ++p) row[p] *= sc;
  }
  __syncthreads();

  // attn = (q @ k^T) * temperature   (per head: 32x32, K=64)
  for (int i = tid; i < 2048; i += 256) {
    int e = i & 31, cc = (i >> 5) & 31, hd = i >> 10;
    const float* qrow = qd[hd * 32 + cc];
    const float* krow = qd[64 + hd * 32 + e];
    float acc = 0.f;
#pragma unroll 8
    for (int p = 0; p < 64; ++p) acc += qrow[p] * krow[p];
    attn[hd][cc][e] = acc * temp[hd];
  }
  __syncthreads();

  // softmax over last dim (32)
  if (tid < 64) {
    float* row = attn[tid >> 5][tid & 31];
    float m = row[0];
    for (int e = 1; e < 32; ++e) m = fmaxf(m, row[e]);
    float s = 0.f;
    for (int e = 0; e < 32; ++e) { float v = expf(row[e] - m); row[e] = v; s += v; }
    float inv = 1.f / s;
    for (int e = 0; e < 32; ++e) row[e] *= inv;
  }
  __syncthreads();

  // out = attn @ v -> ofrag[pix][ch]
  for (int i = tid; i < 4096; i += 256) {
    int p = i & 63, cc = (i >> 6) & 31, hd = i >> 11;
    const float* arow = attn[hd][cc];
    float acc = 0.f;
#pragma unroll 8
    for (int e = 0; e < 32; ++e) acc += arow[e] * qd[128 + hd * 32 + e][p];
    ofrag[p][hd * 32 + cc] = (bf16)acc;
  }
  __syncthreads();

  // proj GEMM (64x64 x 64pix) via WMMA + bias + residual
  const int lane = tid & 31, wave = tid >> 5;
#pragma unroll
  for (int q = 0; q < 2; ++q) {
    int pr = wave * 2 + q;
    int mt = pr >> 2, nt = pr & 3;
    v8f acc = {};
#pragma unroll
    for (int kt = 0; kt < 2; ++kt) {
      v16bf a  = load_a_frag(&Wp[0][0], 64, mt * 16, kt * 32, lane);
      v16bf bb = load_b_frag(&ofrag[0][0], 72, nt * 16, kt * 32, lane);
      acc = wmma_bf16(a, bb, acc);
    }
    int n = nt * 16 + (lane & 15);
    int pix = (h0 + (n >> 3)) * IMGW + w0 + (n & 7);
    int mb = mt * 16 + ((lane >> 4) << 3);
#pragma unroll
    for (int r = 0; r < 8; ++r) {
      int o = mb + r;
      size_t gi = (size_t)(b * 64 + o) * HW_ + pix;
      x2[gi] = acc[r] + projb[o] + x[gi];
    }
  }
}

// ---------------------------------------------------------------------------
// Kernel 4: dwconv3x3(y,340ch) + GELU-gate + pout GEMM (64x170) + residual
// One block per 8x8 tile. 256 threads.
// ---------------------------------------------------------------------------
__global__ void __launch_bounds__(256) ffn2_kernel(
    const bf16* __restrict__ y, const float* __restrict__ dww,
    const float* __restrict__ poutw, const float* __restrict__ x2,
    float* __restrict__ out) {
  __shared__ bf16 yh[340][100];  // 68 KB halo
  __shared__ bf16 g[64][192];    // gated hidden, pixel-major, K zero-padded
  __shared__ bf16 Wo[64][192];   // pout weights, K zero-padded

  const int tid = threadIdx.x;
  const int wi = blockIdx.x;
  const int b = wi >> 10, win = wi & 1023;
  const int h0 = (win >> 5) * 8, w0 = (win & 31) * 8;

  for (int i = tid; i < 340 * 100; i += 256) {
    int c = i / 100, pos = i % 100;
    int gh = h0 + pos / 10 - 1, gw = w0 + pos % 10 - 1;
    bf16 v = (bf16)0.f;
    if ((unsigned)gh < 256u && (unsigned)gw < 256u)
      v = y[(size_t)c * NPIX_ + b * HW_ + gh * IMGW + gw];
    yh[c][pos] = v;
  }
  for (int i = tid; i < 64 * 192; i += 256) {
    int o = i / 192, k = i % 192;
    Wo[o][k] = (k < 170) ? (bf16)poutw[o * 170 + k] : (bf16)0.f;
  }
  for (int i = tid; i < 64 * 22; i += 256) g[i / 22][170 + i % 22] = (bf16)0.f;
  __syncthreads();

  // dwconv both halves + exact GELU gate
  for (int i = tid; i < 170 * 64; i += 256) {
    int cc = i >> 6, p = i & 63, iy = p >> 3, ix = p & 7;
    float a1 = 0.f, a2 = 0.f;
#pragma unroll
    for (int ky = 0; ky < 3; ++ky)
#pragma unroll
      for (int kx = 0; kx < 3; ++kx) {
        int hp = (iy + ky) * 10 + ix + kx;
        a1 += (float)yh[cc][hp]       * dww[cc * 9 + ky * 3 + kx];
        a2 += (float)yh[cc + 170][hp] * dww[(cc + 170) * 9 + ky * 3 + kx];
      }
    float ge = 0.5f * a1 * (1.f + erff(a1 * 0.70710678118f));
    g[p][cc] = (bf16)(ge * a2);
  }
  __syncthreads();

  // pout GEMM (64 x 64pix, K=192 padded) via WMMA + residual
  const int lane = tid & 31, wave = tid >> 5;
#pragma unroll
  for (int q = 0; q < 2; ++q) {
    int pr = wave * 2 + q;
    int mt = pr >> 2, nt = pr & 3;
    v8f acc = {};
#pragma unroll
    for (int kt = 0; kt < 6; ++kt) {
      v16bf a  = load_a_frag(&Wo[0][0], 192, mt * 16, kt * 32, lane);
      v16bf bb = load_b_frag(&g[0][0], 192, nt * 16, kt * 32, lane);
      acc = wmma_bf16(a, bb, acc);
    }
    int n = nt * 16 + (lane & 15);
    int pix = (h0 + (n >> 3)) * IMGW + w0 + (n & 7);
    int mb = mt * 16 + ((lane >> 4) << 3);
#pragma unroll
    for (int r = 0; r < 8; ++r) {
      int o = mb + r;
      size_t gi = (size_t)(b * 64 + o) * HW_ + pix;
      out[gi] = acc[r] + x2[gi];
    }
  }
}

// ---------------------------------------------------------------------------
extern "C" void kernel_launch(void* const* d_in, const int* in_sizes, int n_in,
                              void* d_out, int out_size, void* d_ws, size_t ws_size,
                              hipStream_t stream) {
  const float* x      = (const float*)d_in[0];
  const float* ln1w   = (const float*)d_in[1];
  const float* ln1b   = (const float*)d_in[2];
  const float* qkvw   = (const float*)d_in[3];
  const float* qkvdw  = (const float*)d_in[4];
  const float* temp   = (const float*)d_in[5];
  const float* projw  = (const float*)d_in[6];
  const float* projb  = (const float*)d_in[7];
  const float* ln2w   = (const float*)d_in[8];
  const float* ln2b   = (const float*)d_in[9];
  const float* pinw   = (const float*)d_in[10];
  const float* dww    = (const float*)d_in[11];
  const float* poutw  = (const float*)d_in[12];
  float* outp = (float*)d_out;

  // workspace: x2 (fp32, 64MB) then shared bf16 activation buffer (reused:
  // qkv 192ch then FFN hidden 340ch)
  float* x2 = (float*)d_ws;
  bf16*  buf = (bf16*)((char*)d_ws + (size_t)NPIX_ * 64 * sizeof(float));

  // 1) LN1 + qkv conv1x1 (192x64 GEMM), bf16 out
  ln_conv1x1_kernel<12><<<NPIX_ / 256, 256, 0, stream>>>(x, ln1w, ln1b, qkvw, 192, buf);
  // 2) dwconv + window attention + proj + residual -> x2
  attn_kernel<<<4096, 256, 0, stream>>>(buf, qkvdw, temp, projw, projb, x, x2);
  // 3) LN2 + pin conv1x1 (340x64 GEMM), bf16 out (overwrites qkv buffer)
  ln_conv1x1_kernel<22><<<NPIX_ / 256, 256, 0, stream>>>(x2, ln2w, ln2b, pinw, 340, buf);
  // 4) dwconv + GELU-gate + pout (64x170 GEMM) + residual -> out
  ffn2_kernel<<<4096, 256, 0, stream>>>(buf, dww, poutw, x2, outp);
}
